// DeterministicEncoder_36481452212692
// MI455X (gfx1250) — matrix operand order repaired and checked by reading, not compiled, not run on hardware
//
#include <hip/hip_runtime.h>
#include <hip/hip_bf16.h>

typedef __attribute__((ext_vector_type(16))) _Float16 v16h;
typedef __attribute__((ext_vector_type(8)))  float    v8f;

#define M_CTX   8192
#define N_TGT   8192
#define H_DIM   16
#define OUT_DIM 64

// ---------------------------------------------------------------------------
// Phase 1: MLP encoder  [M,2] -> [M,64], written TRANSPOSED as f16:
//   vT[c][m],  c in [0,64), m in [0,8192)   (1 MiB in d_ws)
// Transposed layout makes the WMMA B-operand (K-major per lane) a contiguous
// 32-byte load. Weights/biases are tiny and broadcast from L2.
// ---------------------------------------------------------------------------
__global__ __launch_bounds__(256) void ncp_mlp_kernel(
    const float* __restrict__ xc, const float* __restrict__ yc,
    const float* __restrict__ W1, const float* __restrict__ b1,
    const float* __restrict__ W2, const float* __restrict__ b2,
    const float* __restrict__ W3, const float* __restrict__ b3,
    _Float16* __restrict__ vT)
{
    int m = blockIdx.x * blockDim.x + threadIdx.x;
    if (m >= M_CTX) return;
    float x = xc[m], y = yc[m];
    float h1[H_DIM], h2[H_DIM];
    #pragma unroll
    for (int j = 0; j < H_DIM; ++j)
        h1[j] = fmaxf(0.f, fmaf(x, W1[j], fmaf(y, W1[H_DIM + j], b1[j])));
    #pragma unroll
    for (int j = 0; j < H_DIM; ++j) {
        float s = b2[j];
        #pragma unroll
        for (int i = 0; i < H_DIM; ++i) s = fmaf(h1[i], W2[i * H_DIM + j], s);
        h2[j] = fmaxf(0.f, s);
    }
    #pragma unroll 4
    for (int c = 0; c < OUT_DIM; ++c) {
        float s = b3[c];
        #pragma unroll
        for (int i = 0; i < H_DIM; ++i) s = fmaf(h2[i], W3[i * OUT_DIM + c], s);
        vT[(size_t)c * M_CTX + m] = (_Float16)s;
    }
}

// ---------------------------------------------------------------------------
// Phase 2: fused Laplace-attention GEMM.
//   out[n, :] = sum_m exp(-|k[m] - q[n]|) * v[m, :]
// One wave32 owns a 16-row q tile x all 64 output columns (4 WMMA C tiles).
// The A matrix (weights) is never materialized in memory: each lane computes
// its 16 f16 A-elements directly, in contiguous k order (k = 16g + e) shared with B:
//   lane group g = lane>>4, row r = lane&15
//   a[e]   = W[r, m0 + 16g + e]       (e = 0..7,  VGPRs 0-3)
//   a[e+8] = W[r, m0 + 16g + 8 + e]   (e = 0..7,  VGPRs 4-7)
// B layout (32x16, K-major): lane n = lane&15, Kbase = m0 + 16g,
//   b[e] = v[Kbase + e, 16*tile + n]  -> 16 contiguous f16 of vT (32B load).
// x_context (32 KB) staged in LDS once per block (320 KB/WGP -> cheap).
// ---------------------------------------------------------------------------
__global__ __launch_bounds__(128) void ncp_attn_kernel(
    const float* __restrict__ xc, const float* __restrict__ xt,
    const _Float16* __restrict__ vT, float* __restrict__ out)
{
    __shared__ float kS[M_CTX];                     // 32 KB
    for (int i = threadIdx.x; i < M_CTX; i += 128) kS[i] = xc[i];
    __syncthreads();

    const int lane = threadIdx.x & 31;              // wave32
    const int wv   = threadIdx.x >> 5;              // 4 waves / block
    const int r    = lane & 15;
    const int g    = lane >> 4;
    const int qbase = (blockIdx.x * 4 + wv) * 16;
    const float q   = xt[qbase + r];

    v8f c0 = {}, c1 = {}, c2 = {}, c3 = {};
    const _Float16* bp0 = vT + (size_t)(r +  0) * M_CTX;
    const _Float16* bp1 = vT + (size_t)(r + 16) * M_CTX;
    const _Float16* bp2 = vT + (size_t)(r + 32) * M_CTX;
    const _Float16* bp3 = vT + (size_t)(r + 48) * M_CTX;

    for (int m0 = 0; m0 < M_CTX; m0 += 32) {
        v16h a;
        const int ka0 = m0 + 16 * g;
        const int ka1 = m0 + 16 * g + 8;
        #pragma unroll
        for (int e = 0; e < 8; ++e) {
            a[e]     = (_Float16)__expf(-fabsf(kS[ka0 + e] - q));
            a[e + 8] = (_Float16)__expf(-fabsf(kS[ka1 + e] - q));
        }
        const int kb = m0 + 16 * g;                 // 32-byte aligned offset
        v16h b0 = *(const v16h*)(bp0 + kb);
        v16h b1 = *(const v16h*)(bp1 + kb);
        v16h b2 = *(const v16h*)(bp2 + kb);
        v16h b3 = *(const v16h*)(bp3 + kb);
        c0 = __builtin_amdgcn_wmma_f32_16x16x32_f16(false, a, false, b0, (short)0, c0, false, false);
        c1 = __builtin_amdgcn_wmma_f32_16x16x32_f16(false, a, false, b1, (short)0, c1, false, false);
        c2 = __builtin_amdgcn_wmma_f32_16x16x32_f16(false, a, false, b2, (short)0, c2, false, false);
        c3 = __builtin_amdgcn_wmma_f32_16x16x32_f16(false, a, false, b3, (short)0, c3, false, false);
    }

    // D layout: VGPR p holds row M = 8g + p, column N = r (per 16x16 f32 C/D).
    #pragma unroll
    for (int p = 0; p < 8; ++p) {
        float* o = out + (size_t)(qbase + 8 * g + p) * OUT_DIM + r;
        o[0]  = c0[p];
        o[16] = c1[p];
        o[32] = c2[p];
        o[48] = c3[p];
    }
}

extern "C" void kernel_launch(void* const* d_in, const int* in_sizes, int n_in,
                              void* d_out, int out_size, void* d_ws, size_t ws_size,
                              hipStream_t stream) {
    (void)in_sizes; (void)n_in; (void)out_size; (void)ws_size;
    const float* xc = (const float*)d_in[0];
    const float* yc = (const float*)d_in[1];
    const float* xt = (const float*)d_in[2];
    const float* W1 = (const float*)d_in[3];
    const float* b1 = (const float*)d_in[4];
    const float* W2 = (const float*)d_in[5];
    const float* b2 = (const float*)d_in[6];
    const float* W3 = (const float*)d_in[7];
    const float* b3 = (const float*)d_in[8];
    float*    out = (float*)d_out;
    _Float16* vT  = (_Float16*)d_ws;   // 64 * 8192 * 2B = 1 MiB scratch

    ncp_mlp_kernel<<<M_CTX / 256, 256, 0, stream>>>(xc, yc, W1, b1, W2, b2, W3, b3, vT);
    ncp_attn_kernel<<<N_TGT / (16 * 4), 128, 0, stream>>>(xc, xt, vT, out);
}
